// SO3Conv_33036888441087
// MI455X (gfx1250) — matrix operand order repaired and checked by reading, not compiled, not run on hardware
//
#include <hip/hip_runtime.h>

// ---------------------------------------------------------------------------
// SO(3) conv for MI455X (gfx1250, wave32).
//  Stage 1: psi[x,y,i] = sum_n w[x,y,n] * D[n,i] / sqrt(1000)
//  Stage 2: per degree l (d=2l+1, block d*d at OFF):
//           out[b,y,v*d+j] = alpha * sum_{x,u} x[b,x,u*d+j] * psi[x,y,u*d+v]
//           == GEMM  M=1024d (rows=(b,j)) x N=256d (cols=(y,v)), K=256d.
//  Precision: split-bf16 (hi + lo residual), 3 x v_wmma_f32_16x16x32_bf16
//             per fp32 MAC tile (f32 accumulate) -> near-fp32 accuracy.
//  Fast path: x / psi pre-packed ONCE into WMMA-fragment-ordered bf16 hi/lo
//  planes in d_ws; stage-2 B fragments are staged through double-buffered
//  LDS with async global->LDS copies shared by the 4 waves of a block.
// ---------------------------------------------------------------------------

#ifndef __has_builtin
#define __has_builtin(x) 0
#endif
#if __has_builtin(__builtin_amdgcn_global_load_async_to_lds_b128) && \
    __has_builtin(__builtin_amdgcn_s_wait_asynccnt)
#define USE_ASYNC_LDS 1
#else
#define USE_ASYNC_LDS 0
#endif

typedef __attribute__((ext_vector_type(16))) __bf16 v16bf;
typedef __attribute__((ext_vector_type(8)))  float  v8f;
typedef int v4i_vs __attribute__((vector_size(16)));               // matches builtin pointee
typedef __attribute__((address_space(1))) v4i_vs as1_v4i;          // global
typedef __attribute__((address_space(3))) v4i_vs as3_v4i;          // LDS

#define F_IN   256
#define F_OUT  256
#define NGRID  1000
#define IRREP  84
#define BATCH  1024

struct Frag { v16bf hi; v16bf lo; };

// native bf16 split: hi = RNE(f), lo = RNE(f - hi)
static __device__ __forceinline__ void frag_set(Frag& fr, int e, float f) {
  __bf16 h = (__bf16)f;
  fr.hi[e] = h;
  fr.lo[e] = (__bf16)(f - (float)h);
}

// c += a*b with hi/lo error compensation (3 bf16 WMMAs, f32 accumulate)
static __device__ __forceinline__ v8f mma3(const v16bf& ah, const v16bf& al,
                                           const v16bf& bh, const v16bf& bl,
                                           v8f c) {
  c = __builtin_amdgcn_wmma_f32_16x16x32_bf16(false, ah, false, bl, (short)0, c, false, false);
  c = __builtin_amdgcn_wmma_f32_16x16x32_bf16(false, al, false, bh, (short)0, c, false, false);
  c = __builtin_amdgcn_wmma_f32_16x16x32_bf16(false, ah, false, bh, (short)0, c, false, false);
  return c;
}

// --- packed-workspace geometry (bf16 element offsets) ----------------------
// Apack_l : fragment order ((mt16*KS + ks)*32 + lane)*32 + plane*16 + e
// Bpack_l : fragment order ((nt16*KS + ks)*32 + lane)*32 + plane*16 + e
static constexpr size_t AP_OFF[4] = {0, 524288, 5242880, 18350080};
static constexpr size_t BP_OFF[4] = {0, 131072, 1310720, 4587520};
#define AP_ELEMS 44040192   // sum_l 1024d*256d*2
#define BP_ELEMS 11010048   // sum_l 256d*256d*2

// ---------------------------------------------------------------------------
// Stage 0 (fast path): pack x into A-fragment bf16 hi/lo planes.
// ---------------------------------------------------------------------------
template <int D_, int OFF>
__global__ void __launch_bounds__(256) pack_x(const float* __restrict__ x,
                                              unsigned short* __restrict__ apack) {
  constexpr int KS = (F_IN * D_) / 32;
  const int tid  = blockIdx.x * blockDim.x + threadIdx.x;
  const int lane = tid & 31;
  const int frag = tid >> 5;                 // mt16*KS + ks
  const int mt16 = frag / KS;
  const int ks   = frag % KS;
  const int lr   = lane & 15;
  const int kh   = lane >> 4;
  const int r    = mt16 * 16 + lr;           // GEMM row = (b, j)
  const int b    = r / D_, j = r % D_;
  const float* __restrict__ base = x + (size_t)b * (F_IN * IRREP) + OFF + j;

  Frag fr;
#pragma unroll
  for (int e = 0; e < 16; ++e) {
    int k  = ks * 32 + kh * 16 + e;
    int xi = k / D_, u = k % D_;
    frag_set(fr, e, base[xi * IRREP + u * D_]);
  }
  unsigned short* dst = apack + ((size_t)frag * 32 + lane) * 32;
  *(v16bf*)(dst)      = fr.hi;
  *(v16bf*)(dst + 16) = fr.lo;
}

// ---------------------------------------------------------------------------
// Stage 1: psi = (w as 65536x1000) @ D (1000x84) * 1/sqrt(1000)
// Guards: address clamp + value select (no EXEC churn).
// ---------------------------------------------------------------------------
template <bool PACKED>
__global__ void __launch_bounds__(128) psi_gemm(const float* __restrict__ w,
                                                const float* __restrict__ Dm,
                                                float* __restrict__ psiF32,
                                                unsigned short* __restrict__ bpack) {
  const int lane = threadIdx.x & 31;
  const int wid  = blockIdx.x * (blockDim.x >> 5) + (threadIdx.x >> 5);
  const int kh   = lane >> 4;
  const int lr   = lane & 15;
  const int m    = wid * 16 + lr;
  const float* __restrict__ wrow = w + (size_t)m * NGRID;

  const v8f vzero = {};
  v8f acc[6];
#pragma unroll
  for (int nt = 0; nt < 6; ++nt) acc[nt] = vzero;

  for (int k0 = 0; k0 < NGRID; k0 += 32) {
    Frag a;
#pragma unroll
    for (int e = 0; e < 16; ++e) {
      int k  = k0 + kh * 16 + e;
      int kc = (k < NGRID) ? k : (NGRID - 1);
      float f = wrow[kc];
      frag_set(a, e, (k < NGRID) ? f : 0.0f);
    }
#pragma unroll
    for (int nt = 0; nt < 6; ++nt) {
      const int n  = nt * 16 + lr;
      const int nc = (n < IRREP) ? n : (IRREP - 1);
      Frag bf;
#pragma unroll
      for (int e = 0; e < 16; ++e) {
        int k  = k0 + kh * 16 + e;
        int kc = (k < NGRID) ? k : (NGRID - 1);
        float f = Dm[(size_t)kc * IRREP + nc];
        frag_set(bf, e, (k < NGRID && n < IRREP) ? f : 0.0f);
      }
      acc[nt] = mma3(a.hi, a.lo, bf.hi, bf.lo, acc[nt]);
    }
  }

  const float scale = 0.0316227766016838f; // 1/sqrt(1000)
#pragma unroll
  for (int nt = 0; nt < 6; ++nt) {
    const int n = nt * 16 + lr;
    if (n >= IRREP) continue;
#pragma unroll
    for (int r = 0; r < 8; ++r) {
      const int   mrow = wid * 16 + r + 8 * kh;
      const float val  = acc[nt][r] * scale;
      if constexpr (!PACKED) {
        psiF32[(size_t)mrow * IRREP + n] = val;
      } else {
        // scatter into stage-2 B-fragment layout:
        //   B_l[k = x*d + u][c = y*d + v] = psi[x, y, OFF + u*d + v]
        int d, off; size_t bpo;
        if      (n < 1)  { d = 1; off = 0;  bpo = BP_OFF[0]; }
        else if (n < 10) { d = 3; off = 1;  bpo = BP_OFF[1]; }
        else if (n < 35) { d = 5; off = 10; bpo = BP_OFF[2]; }
        else             { d = 7; off = 35; bpo = BP_OFF[3]; }
        const int xi  = mrow >> 8;
        const int y   = mrow & 255;
        const int rel = n - off;
        const int u   = rel / d, v = rel % d;
        const int k   = xi * d + u;
        const int c   = y * d + v;
        const int KSl = 8 * d;
        const int fB  = (c >> 4) * KSl + (k >> 5);
        const int lb  = ((k >> 4) & 1) * 16 + (c & 15);
        const int e   = k & 15;
        union { __bf16 b; unsigned short u16; } ch, cl;
        ch.b = (__bf16)val;
        cl.b = (__bf16)(val - (float)ch.b);
        size_t o = bpo + ((size_t)fB * 32 + lb) * 32 + e;
        bpack[o]      = ch.u16;
        bpack[o + 16] = cl.u16;
      }
    }
  }
}

// ---------------------------------------------------------------------------
// Stage 2 (fast path): fragment GEMM with async double-buffered LDS staging.
// Block = 4 waves sharing one N-tile (nt): B fragments (8KB/K-step) are
// copied global->LDS once per block (async when available) and consumed by
// all 4 waves; A fragments are per-wave direct b128 loads.  24 WMMAs/step.
// ---------------------------------------------------------------------------
template <int D_, int OFF>
__global__ void __launch_bounds__(128) gemm_packed(const unsigned short* __restrict__ apack,
                                                   const unsigned short* __restrict__ bpack,
                                                   float* __restrict__ out) {
  constexpr int KS   = (F_IN * D_) / 32;
  constexpr int MT32 = (BATCH * D_) / 32;          // 32d, divisible by 4
  __shared__ alignas(16) unsigned short sB[2][4096];   // 2 bufs x 4 frags x 2KB

  const int tid  = threadIdx.x;
  const int lane = tid & 31;
  const int wid  = blockIdx.x * 4 + (tid >> 5);
  const int mt   = wid % MT32;                     // per-wave M tile
  const int nt   = wid / MT32;                     // shared by the block
  const int kh   = lane >> 4;
  const int lr   = lane & 15;

  // stage B fragments (4 x 2KB) for K-step ks into LDS buffer buf
  auto stage_b = [&](int ks, int buf) {
#pragma unroll
    for (int i = 0; i < 4; ++i) {
      const unsigned short* g = bpack + ((size_t)((nt * 4 + i) * KS + ks)) * 1024 + tid * 8;
      unsigned short* l = &sB[buf][i * 1024 + tid * 8];
#if USE_ASYNC_LDS
      __builtin_amdgcn_global_load_async_to_lds_b128(
          (as1_v4i*)(unsigned long long)(__SIZE_TYPE__)g,
          (as3_v4i*)(unsigned)(__SIZE_TYPE__)l, 0, 0);
#else
      *(uint4*)l = *(const uint4*)g;               // global_load_b128 + ds_store_b128
#endif
    }
  };

  const unsigned short* aP[2];
#pragma unroll
  for (int mh = 0; mh < 2; ++mh)
    aP[mh] = apack + ((size_t)((mt * 2 + mh) * KS) * 32 + lane) * 32;

  const v8f vzero = {};
  v8f acc[2][4];
#pragma unroll
  for (int mh = 0; mh < 2; ++mh)
#pragma unroll
    for (int n4 = 0; n4 < 4; ++n4) acc[mh][n4] = vzero;

  stage_b(0, 0);

  for (int ks = 0; ks < KS; ++ks) {
    const int cur = ks & 1;
#if USE_ASYNC_LDS
    __builtin_amdgcn_s_wait_asynccnt(0);           // this wave's copies done
#endif
    __syncthreads();                               // all waves' copies visible
    if (ks + 1 < KS) stage_b(ks + 1, cur ^ 1);     // overlap next copy w/ WMMA

    __builtin_prefetch(aP[0] + 2048, 0, 3);        // A two K-steps ahead
    v16bf ah[2], al[2];
#pragma unroll
    for (int mh = 0; mh < 2; ++mh) {
      ah[mh] = *(const v16bf*)(aP[mh]);
      al[mh] = *(const v16bf*)(aP[mh] + 16);
      aP[mh] += 1024;                              // 32 lanes * 32 elems
    }
#pragma unroll
    for (int n4 = 0; n4 < 4; ++n4) {
      const v16bf* bp = (const v16bf*)(&sB[cur][n4 * 1024 + lane * 32]);
      v16bf bh = bp[0];
      v16bf bl = bp[1];
#pragma unroll
      for (int mh = 0; mh < 2; ++mh)
        acc[mh][n4] = mma3(ah[mh], al[mh], bh, bl, acc[mh][n4]);
    }
  }

  const float alpha = (D_ == 1) ? 0.0625f
                    : (D_ == 3) ? 0.0360843918243516f
                    : (D_ == 5) ? 0.0279508497187474f
                                : 0.0236227795630767f;
#pragma unroll
  for (int mh = 0; mh < 2; ++mh) {
#pragma unroll
    for (int n4 = 0; n4 < 4; ++n4) {
      int col = nt * 64 + n4 * 16 + lr;
      int y = col / D_, v = col % D_;
#pragma unroll
      for (int r = 0; r < 8; ++r) {
        int row = mt * 32 + mh * 16 + r + 8 * kh;
        int b = row / D_, j = row % D_;
        out[((size_t)b * F_OUT + y) * IRREP + OFF + v * D_ + j] =
            acc[mh][n4][r] * alpha;
      }
    }
  }
}

// ---------------------------------------------------------------------------
// Stage 2 (fallback, small d_ws): direct gather + in-loop bf16 split.
// ---------------------------------------------------------------------------
template <int D_, int OFF>
__global__ void __launch_bounds__(128) gemm_direct(const float* __restrict__ x,
                                                   const float* __restrict__ psi,
                                                   float* __restrict__ out) {
  constexpr int K    = F_IN * D_;
  constexpr int NT64 = (F_OUT * D_) / 64;
  const int lane  = threadIdx.x & 31;
  const int wid   = blockIdx.x * (blockDim.x >> 5) + (threadIdx.x >> 5);
  const int mt    = wid / NT64;
  const int nt    = wid % NT64;
  const int kh    = lane >> 4;
  const int lr    = lane & 15;

  const float* __restrict__ aBase[2];
#pragma unroll
  for (int mh = 0; mh < 2; ++mh) {
    int r = mt * 32 + mh * 16 + lr;
    int b = r / D_, j = r % D_;
    aBase[mh] = x + (size_t)b * (F_IN * IRREP) + OFF + j;
  }
  const float* __restrict__ bBase[4];
#pragma unroll
  for (int n4 = 0; n4 < 4; ++n4) {
    int c = nt * 64 + n4 * 16 + lr;
    int y = c / D_, v = c % D_;
    bBase[n4] = psi + (size_t)y * IRREP + OFF + v;
  }

  const v8f vzero = {};
  v8f acc[2][4];
#pragma unroll
  for (int mh = 0; mh < 2; ++mh)
#pragma unroll
    for (int n4 = 0; n4 < 4; ++n4) acc[mh][n4] = vzero;

  for (int k0 = 0; k0 < K; k0 += 32) {
    Frag a[2];
#pragma unroll
    for (int mh = 0; mh < 2; ++mh) {
#pragma unroll
      for (int e = 0; e < 16; ++e) {
        int k  = k0 + kh * 16 + e;
        int xi = k / D_, u = k % D_;
        frag_set(a[mh], e, aBase[mh][xi * IRREP + u * D_]);
      }
    }
#pragma unroll
    for (int n4 = 0; n4 < 4; ++n4) {
      Frag b;
#pragma unroll
      for (int e = 0; e < 16; ++e) {
        int k  = k0 + kh * 16 + e;
        int xi = k / D_, u = k % D_;
        frag_set(b, e, bBase[n4][(size_t)xi * (F_OUT * IRREP) + u * D_]);
      }
#pragma unroll
      for (int mh = 0; mh < 2; ++mh)
        acc[mh][n4] = mma3(a[mh].hi, a[mh].lo, b.hi, b.lo, acc[mh][n4]);
    }
  }

  const float alpha = (D_ == 1) ? 0.0625f
                    : (D_ == 3) ? 0.0360843918243516f
                    : (D_ == 5) ? 0.0279508497187474f
                                : 0.0236227795630767f;
#pragma unroll
  for (int mh = 0; mh < 2; ++mh) {
#pragma unroll
    for (int n4 = 0; n4 < 4; ++n4) {
      int col = nt * 64 + n4 * 16 + lr;
      int y = col / D_, v = col % D_;
#pragma unroll
      for (int r = 0; r < 8; ++r) {
        int row = mt * 32 + mh * 16 + r + 8 * kh;
        int b = row / D_, j = row % D_;
        out[((size_t)b * F_OUT + y) * IRREP + OFF + v * D_ + j] =
            acc[mh][n4][r] * alpha;
      }
    }
  }
}

extern "C" void kernel_launch(void* const* d_in, const int* in_sizes, int n_in,
                              void* d_out, int out_size, void* d_ws, size_t ws_size,
                              hipStream_t stream) {
  (void)in_sizes; (void)n_in; (void)out_size;
  const float* x  = (const float*)d_in[0];   // (1024, 256, 84)
  const float* w  = (const float*)d_in[1];   // (256, 256, 1000)
  const float* Dm = (const float*)d_in[2];   // (1000, 84)
  float* out = (float*)d_out;                // (1024, 256, 84)

  const size_t WS_FULL = (size_t)(AP_ELEMS + BP_ELEMS) * 2;  // ~105 MB

  if (ws_size >= WS_FULL) {
    unsigned short* apack = (unsigned short*)d_ws;
    unsigned short* bpack = apack + AP_ELEMS;
    pack_x<1,  0><<<  64, 256, 0, stream>>>(x, apack + AP_OFF[0]);
    pack_x<3,  1><<< 576, 256, 0, stream>>>(x, apack + AP_OFF[1]);
    pack_x<5, 10><<<1600, 256, 0, stream>>>(x, apack + AP_OFF[2]);
    pack_x<7, 35><<<3136, 256, 0, stream>>>(x, apack + AP_OFF[3]);
    psi_gemm<true><<<1024, 128, 0, stream>>>(w, Dm, nullptr, bpack);
    gemm_packed<1,  0><<<  32, 128, 0, stream>>>(apack + AP_OFF[0], bpack + BP_OFF[0], out);
    gemm_packed<3,  1><<< 288, 128, 0, stream>>>(apack + AP_OFF[1], bpack + BP_OFF[1], out);
    gemm_packed<5, 10><<< 800, 128, 0, stream>>>(apack + AP_OFF[2], bpack + BP_OFF[2], out);
    gemm_packed<7, 35><<<1568, 128, 0, stream>>>(apack + AP_OFF[3], bpack + BP_OFF[3], out);
  } else {
    float* psi = (float*)d_ws;
    psi_gemm<false><<<1024, 128, 0, stream>>>(w, Dm, psi, nullptr);
    gemm_direct<1,  0><<<  32, 128, 0, stream>>>(x, psi, out);
    gemm_direct<3,  1><<< 288, 128, 0, stream>>>(x, psi, out);
    gemm_direct<5, 10><<< 800, 128, 0, stream>>>(x, psi, out);
    gemm_direct<7, 35><<<1568, 128, 0, stream>>>(x, psi, out);
  }
}
